// SineRnn_37804302139695
// MI455X (gfx1250) — matrix operand order repaired and checked by reading, not compiled, not run on hardware
//
#include <hip/hip_runtime.h>
#include <hip/hip_fp16.h>

// ---------------------------------------------------------------------------
// 2-layer LSTM (H=512, B=256) persistent-kernel implementation for gfx1250.
//  - f16 WMMA 16x16x32, f32 accumulate (v_wmma_f32_16x16x32_f16)
//  - cell state c1/c2 kept in VGPRs across all timesteps
//  - weights packed once into WMMA-fragment-major f16 (L2 resident, 6 MB)
//  - A-tiles (h state) staged per-block into LDS via async global->LDS copies
//    (ASYNCcnt path) and consumed with ds_load_b128 — 8 waves/WG share one tile
//  - 2 grid barriers per timestep, h double-buffered by step parity
// ---------------------------------------------------------------------------

typedef __attribute__((ext_vector_type(16))) _Float16 v16h;
typedef __attribute__((ext_vector_type(8)))  _Float16 v8h;
typedef __attribute__((ext_vector_type(8)))  float    v8f;

// Types for the async global->LDS builtin:
//   param 1: AS1 (global, printed "__device__") pointer to int __vector(4)
//   param 2: AS3 (LDS) pointer to the same 16-byte vector type
typedef int b128_t __attribute__((vector_size(16)));
typedef b128_t __attribute__((address_space(1))) b128_g;
typedef b128_t __attribute__((address_space(3))) b128_l;

#define NBLK 64      // persistent grid: 64 WGs x 256 thr = 512 waves (wave32)
#define TPB  256
#define HSZ  512     // hidden
#define BSZ  256     // batch
#define LDSTRIDE 1032   // halfs per LDS A-tile row (1024 + 8 pad -> bank rotate)

#if __has_builtin(__builtin_amdgcn_global_load_async_to_lds_b128)
#define HAVE_ASYNC_LDS 1
#else
#define HAVE_ASYNC_LDS 0
#endif

__device__ __forceinline__ float sigm(float x) { return 1.0f / (1.0f + __expf(-x)); }
__device__ __forceinline__ float tanh_fast(float x) {
    float e = __expf(-2.0f * fabsf(x));
    float t = (1.0f - e) / (1.0f + e);
    return copysignf(t, x);
}

// 16-byte global -> LDS copy (async if the toolchain exposes the builtin).
__device__ __forceinline__ void cp16(_Float16* lds_dst, const _Float16* gsrc) {
#if HAVE_ASYNC_LDS
    b128_g* gp = (b128_g*)(uintptr_t)gsrc;                    // AS1, value = 64-bit VA
    b128_l* lp = (b128_l*)(unsigned)(uintptr_t)lds_dst;       // AS3, value = LDS offset
    __builtin_amdgcn_global_load_async_to_lds_b128(gp, lp, 0, 0);
#else
    *(v8h*)lds_dst = *(const v8h*)gsrc;
#endif
}

__device__ __forceinline__ void cp_wait() {
#if HAVE_ASYNC_LDS
#if __has_builtin(__builtin_amdgcn_s_wait_asynccnt)
    __builtin_amdgcn_s_wait_asynccnt(0);
#else
    asm volatile("s_wait_asynccnt 0" ::: "memory");
#endif
#endif
    __syncthreads();
}

// Sense-reversal grid barrier (all NBLK blocks co-resident by construction).
__device__ __forceinline__ void grid_sync(unsigned* cnt, volatile unsigned* gen) {
    __threadfence();
    __syncthreads();
    if (threadIdx.x == 0) {
        unsigned g = *gen;
        if (atomicAdd(cnt, 1u) == NBLK - 1u) {
            *cnt = 0u;
            __threadfence();
            atomicAdd((unsigned*)gen, 1u);
        } else {
            while (*gen == g) { __builtin_amdgcn_s_sleep(1); }
        }
    }
    __syncthreads();
    __threadfence();
}

// ---------------------------------------------------------------------------
// Weight packing: f32 [K,2048] row-major  ->  f16 fragment-major.
// B-fragment layout used here for v_wmma_f32_16x16x32_f16 (32x16 B tile, A uses same k order):
//   lane l holds column (l&15); element e holds K row = 32*k + 16*(l>>4) + e.
// Packed offset (halfs): ((n*KT + k)*32 + lane)*16 + e
// ---------------------------------------------------------------------------
__global__ void pack_w1(const float* __restrict__ Wh1, _Float16* __restrict__ W1p) {
    int idx = blockIdx.x * blockDim.x + threadIdx.x;   // 128 ntiles * 16 ktiles * 32 lanes
    int n = idx >> 9;
    int k = (idx >> 5) & 15;
    int l = idx & 31;
    int col = (n << 4) + (l & 15);
    _Float16* out = W1p + (size_t)((n * 16 + k) * 512 + l * 16);
    int kr0 = (k << 5) + ((l >> 4) << 4);
#pragma unroll
    for (int e = 0; e < 16; e++)
        out[e] = (_Float16)Wh1[(size_t)(kr0 + e) * 2048 + col];
}

__global__ void pack_w2(const float* __restrict__ Wx2, const float* __restrict__ Wh2,
                        _Float16* __restrict__ W2p) {
    int idx = blockIdx.x * blockDim.x + threadIdx.x;   // 128 ntiles * 32 ktiles * 32 lanes
    int n = idx >> 10;
    int k = (idx >> 5) & 31;
    int l = idx & 31;
    int col = (n << 4) + (l & 15);
    _Float16* out = W2p + (size_t)((n * 32 + k) * 512 + l * 16);
    int kr0 = (k << 5) + ((l >> 4) << 4);
#pragma unroll
    for (int e = 0; e < 16; e++) {
        int kr = kr0 + e;
        float v = (kr < HSZ) ? Wx2[(size_t)kr * 2048 + col]
                             : Wh2[(size_t)(kr - HSZ) * 2048 + col];
        out[e] = (_Float16)v;
    }
}

// ---------------------------------------------------------------------------
// Persistent timestep loop.
//   hcat buffers: [256 rows x 1024 cols] f16; cols 0..511 = h1, 512..1023 = h2.
//   Wave job: mt = wave>>5 (16 M-tiles, block-uniform), ut = wave&31 (32 tiles).
//   Gate tiles for job: N-tiles {g*32+ut}, g=0..3  (gates packed [i,f,o,g]).
// ---------------------------------------------------------------------------
__global__ void __launch_bounds__(TPB) lstm_persistent(
    const float* __restrict__ seq, const float* __restrict__ Wx1,
    const float* __restrict__ bx1, const float* __restrict__ bh1,
    const float* __restrict__ bx2, const float* __restrict__ bh2,
    const float* __restrict__ Wl,  const float* __restrict__ bl,
    const _Float16* __restrict__ W1p, const _Float16* __restrict__ W2p,
    _Float16* __restrict__ hcat0, _Float16* __restrict__ hcat1,
    float* __restrict__ obuf, unsigned* __restrict__ bar,
    float* __restrict__ out, int T, int TP)
{
    __shared__ _Float16 Atile[16 * LDSTRIDE];   // 33 KB: block's 16-row A tile

    const int lane = threadIdx.x & 31;
    const int wave = (blockIdx.x << 3) + (threadIdx.x >> 5);  // 0..511
    const int mt   = wave >> 5;                                // 0..15 (block-uniform)
    const int ut   = wave & 31;                                // 0..31
    const int lh   = lane >> 4;                                // K-half / C-row-half
    const int ln   = lane & 15;                                // A row / B,C column
    const float blv = bl[0];

    unsigned* cnt = bar;
    volatile unsigned* gen = bar + 1;

    // Per-wave constants across all timesteps.
    const int ucol = (ut << 4) + ln;            // unit column 0..511
    float w1x[4], b1c[4], b2c[4];
#pragma unroll
    for (int g = 0; g < 4; g++) {
        int cg = g * HSZ + ucol;
        w1x[g] = Wx1[cg];
        b1c[g] = bx1[cg] + bh1[cg];
        b2c[g] = bx2[cg] + bh2[cg];
    }
    const float wlv = Wl[ucol];
    const int rowbase = (mt << 4) + (lh << 3);  // C/D fragment row base

    v8f c1 = {};  // layer-1 cell state, resident in registers for all steps
    v8f c2 = {};  // layer-2 cell state

    for (int s = 0; s < TP; s++) {
        const _Float16* hprev = (s & 1) ? hcat0 : hcat1;  // buf[(s-1)&1]
        _Float16*       hcur  = (s & 1) ? hcat1 : hcat0;  // buf[s&1]
        float* oacc  = obuf + ((s & 1) << 8);
        float* oprev = obuf + (((s + 1) & 1) << 8);

        // Housekeeping (within phase A, separated from phase B by a barrier):
        if (blockIdx.x == 0 && s > 0) {
            int b = threadIdx.x;                       // 256 threads, 256 rows
            out[(size_t)b * TP + (s - 1)] = oprev[b] + blv;
        }
        if (blockIdx.x == 1) { oacc[threadIdx.x] = 0.0f; }

        // ---- Phase A: stage h1_prev tile (16 x 512 f16) into LDS ----------
        {
            const _Float16* base = hprev + (size_t)(mt << 4) * 1024;
#pragma unroll
            for (int i = 0; i < 4; i++) {              // 1024 16B-chunks / 256 thr
                int c   = threadIdx.x + (i << 8);
                int row = c >> 6;
                int off = (c & 63) << 3;
                cp16(&Atile[row * LDSTRIDE + off], base + (size_t)row * 1024 + off);
            }
            cp_wait();
        }

        // ---- Phase A: z1 = h1_prev @ Wh1 (K=512), then gate activations ----
        v8f acc[4] = {};
        {
#pragma unroll 4
            for (int k = 0; k < 16; k++) {
                __builtin_prefetch(W1p + (size_t)((ut * 16 + ((k + 1) & 15)) * 512), 0, 1);
                const _Float16* ap = &Atile[ln * LDSTRIDE + (k << 5) + (lh << 4)];
                v8h alo = *(const v8h*)(ap);
                v8h ahi = *(const v8h*)(ap + 8);
                v16h a = __builtin_shufflevector(alo, ahi,
                         0,1,2,3,4,5,6,7,8,9,10,11,12,13,14,15);
#pragma unroll
                for (int g = 0; g < 4; g++) {
                    const _Float16* bp = W1p + (size_t)(((g * 32 + ut) * 16 + k) * 512)
                                             + (lane << 4);
                    v16h bf = *(const v16h*)bp;
                    acc[g] = __builtin_amdgcn_wmma_f32_16x16x32_f16(
                                 false, a, false, bf, (short)0, acc[g], false, false);
                }
            }
        }
#pragma unroll
        for (int r = 0; r < 8; r++) {
            int row = rowbase + r;
            float xv = (s < T) ? seq[(size_t)row * T + s] : (oprev[row] + blv);
            float zi = acc[0][r] + xv * w1x[0] + b1c[0];
            float zf = acc[1][r] + xv * w1x[1] + b1c[1];
            float zo = acc[2][r] + xv * w1x[2] + b1c[2];
            float zg = acc[3][r] + xv * w1x[3] + b1c[3];
            float ig = sigm(zi), fg = sigm(zf), og = sigm(zo), gg = tanh_fast(zg);
            float cn = fg * c1[r] + ig * gg;
            c1[r] = cn;
            float hn = og * tanh_fast(cn);
            hcur[(size_t)row * 1024 + ucol] = (_Float16)hn;   // h1 region
        }

        grid_sync(cnt, gen);   // entry __syncthreads also closes out Atile reads

        // ---- Phase B: stage [h1_new | h2_prev] tile (16 x 1024 f16) -------
        {
#pragma unroll
            for (int i = 0; i < 8; i++) {              // 2048 16B-chunks / 256 thr
                int c   = threadIdx.x + (i << 8);
                int row = c >> 7;
                int off = (c & 127) << 3;
                const _Float16* src = ((off < HSZ) ? hcur : hprev)
                                    + (size_t)((mt << 4) + row) * 1024 + off;
                cp16(&Atile[row * LDSTRIDE + off], src);
            }
            cp_wait();
        }

        // ---- Phase B: z2 = [h1_new | h2_prev] @ [Wx2;Wh2]  (K=1024) -------
        acc[0] = (v8f){}; acc[1] = (v8f){}; acc[2] = (v8f){}; acc[3] = (v8f){};
        {
#pragma unroll 4
            for (int k = 0; k < 32; k++) {
                __builtin_prefetch(W2p + (size_t)((ut * 32 + ((k + 1) & 31)) * 512), 0, 1);
                const _Float16* ap = &Atile[ln * LDSTRIDE + (k << 5) + (lh << 4)];
                v8h alo = *(const v8h*)(ap);
                v8h ahi = *(const v8h*)(ap + 8);
                v16h a = __builtin_shufflevector(alo, ahi,
                         0,1,2,3,4,5,6,7,8,9,10,11,12,13,14,15);
#pragma unroll
                for (int g = 0; g < 4; g++) {
                    const _Float16* bp = W2p + (size_t)(((g * 32 + ut) * 32 + k) * 512)
                                             + (lane << 4);
                    v16h bf = *(const v16h*)bp;
                    acc[g] = __builtin_amdgcn_wmma_f32_16x16x32_f16(
                                 false, a, false, bf, (short)0, acc[g], false, false);
                }
            }
        }
#pragma unroll
        for (int r = 0; r < 8; r++) {
            int row = rowbase + r;
            float zi = acc[0][r] + b2c[0];
            float zf = acc[1][r] + b2c[1];
            float zo = acc[2][r] + b2c[2];
            float zg = acc[3][r] + b2c[3];
            float ig = sigm(zi), fg = sigm(zf), og = sigm(zo), gg = tanh_fast(zg);
            float cn = fg * c2[r] + ig * gg;
            c2[r] = cn;
            float hn = og * tanh_fast(cn);
            hcur[(size_t)row * 1024 + HSZ + ucol] = (_Float16)hn;  // h2 region
            // Output head partial: o[row] += sum_cols h2[row,col]*Wl[col]
            float p = hn * wlv;
            p += __shfl_xor(p, 1, 32);
            p += __shfl_xor(p, 2, 32);
            p += __shfl_xor(p, 4, 32);
            p += __shfl_xor(p, 8, 32);
            if (ln == 0) atomicAdd(&oacc[row], p);
        }

        grid_sync(cnt, gen);
    }

    // Final timestep's output.
    if (blockIdx.x == 0) {
        int b = threadIdx.x;
        const float* olast = obuf + (((TP + 1) & 1) << 8);  // (TP-1)&1
        out[(size_t)b * TP + (TP - 1)] = olast[b] + blv;
    }
}

// ---------------------------------------------------------------------------
extern "C" void kernel_launch(void* const* d_in, const int* in_sizes, int n_in,
                              void* d_out, int out_size, void* d_ws, size_t ws_size,
                              hipStream_t stream) {
    const float* seq = (const float*)d_in[0];
    // d_in[1] = predict (scalar, derived from out_size instead)
    const float* Wx1 = (const float*)d_in[2];
    const float* bx1 = (const float*)d_in[3];
    const float* Wh1 = (const float*)d_in[4];
    const float* bh1 = (const float*)d_in[5];
    const float* Wx2 = (const float*)d_in[6];
    const float* bx2 = (const float*)d_in[7];
    const float* Wh2 = (const float*)d_in[8];
    const float* bh2 = (const float*)d_in[9];
    const float* Wl  = (const float*)d_in[10];
    const float* bl  = (const float*)d_in[11];
    float* out = (float*)d_out;

    int T  = in_sizes[0] / BSZ;   // 512
    int TP = out_size / BSZ;      // T + predict

    // Workspace layout (all offsets 256B-aligned):
    char* ws = (char*)d_ws;
    _Float16* W1p   = (_Float16*)(ws);                    // 2 MB packed Wh1
    _Float16* W2p   = (_Float16*)(ws + (2u << 20));       // 4 MB packed [Wx2;Wh2]
    char* zbase     = ws + (6u << 20);
    _Float16* hcat0 = (_Float16*)(zbase);                 // 512 KB
    _Float16* hcat1 = (_Float16*)(zbase + 524288);        // 512 KB
    float*    obuf  = (float*)(zbase + 2 * 524288);       // 2 KB (double-buffered o)
    unsigned* bar   = (unsigned*)(zbase + 2 * 524288 + 4096);
    size_t zbytes   = 2 * 524288 + 4096 + 256;

    (void)hipMemsetAsync(zbase, 0, zbytes, stream);       // h=0, c via regs, barrier=0
    pack_w1<<<256, 256, 0, stream>>>(Wh1, W1p);
    pack_w2<<<512, 256, 0, stream>>>(Wx2, Wh2, W2p);
    lstm_persistent<<<NBLK, TPB, 0, stream>>>(
        seq, Wx1, bx1, bh1, bx2, bh2, Wl, bl,
        W1p, W2p, hcat0, hcat1, obuf, bar, out, T, TP);
}
